// KeypointGNN_37460704756040
// MI455X (gfx1250) — matrix operand order repaired and checked by reading, not compiled, not run on hardware
//
#include <hip/hip_runtime.h>

#define NN 50000
#define EE 1600000
#define GG 512
#define ET (EE + NN)   // edges + self loops = 1,650,000
#define FD 128

typedef __attribute__((ext_vector_type(16))) _Float16 v16h;
typedef __attribute__((ext_vector_type(8)))  _Float16 v8h;
typedef __attribute__((ext_vector_type(8)))  float    v8f;

// order-preserving float <-> uint mapping for atomicMax on floats
__device__ __forceinline__ unsigned fkey(float x) {
  unsigned u = __float_as_uint(x);
  return (u & 0x80000000u) ? ~u : (u | 0x80000000u);
}
__device__ __forceinline__ float funkey(unsigned k) {
  return (k & 0x80000000u) ? __uint_as_float(k & 0x7fffffffu) : __uint_as_float(~k);
}

// ---------------- layer 0 linear: h = x @ W0 (K=4, plain VALU) ----------------
__global__ void k_lin0(const float* __restrict__ x, const float* __restrict__ W0,
                       float* __restrict__ h) {
  int idx = blockIdx.x * blockDim.x + threadIdx.x;
  if (idx >= NN * FD) return;
  int n = idx >> 7, j = idx & 127;
  const float* xr = x + n * 4;
  h[idx] = xr[0] * W0[j] + xr[1] * W0[FD + j] + xr[2] * W0[2 * FD + j] + xr[3] * W0[3 * FD + j];
}

// ------------- convert W [in=128, out=128] f32 -> transposed half Wt[out][in] -------------
__global__ void k_wT_half(const float* __restrict__ W, _Float16* __restrict__ Wt) {
  int idx = blockIdx.x * blockDim.x + threadIdx.x;
  if (idx >= FD * FD) return;
  int k = idx >> 7, j = idx & 127;
  Wt[j * FD + k] = (_Float16)W[k * FD + j];
}

// ------------- WMMA GEMM: C[M,128] = A[M,128] (f16) * W[128,128], Wt = W^T (f16) -------------
// one wave per 16x16 output tile; 8 waves/block cover all 128 output columns
__global__ void k_wmma_gemm(const _Float16* __restrict__ Ah,
                            const _Float16* __restrict__ Wt,
                            float* __restrict__ Cf) {
  const int lane = threadIdx.x & 31;
  const int wave = threadIdx.x >> 5;
  const int row0 = blockIdx.x * 16;
  const int col0 = wave * 16;
  const int m = lane & 15;
  const int g = lane >> 4;
  v8f acc = {};
  const _Float16* arow = Ah + (size_t)(row0 + m) * FD;
  const _Float16* brow = Wt + (size_t)(col0 + m) * FD;
#pragma unroll
  for (int kk = 0; kk < FD; kk += 32) {
    // A 16x32 f16 tile: lane (m,g) holds K = {8g..8g+7} and {16+8g..16+8g+7}
    v8h a0 = *(const v8h*)(arow + kk + 8 * g);
    v8h a1 = *(const v8h*)(arow + kk + 16 + 8 * g);
    // B 32x16 f16 tile: lane (n=m, g) holds K = {8g..8g+7} and {16+8g..16+8g+7} in Wt row
    v8h b0 = *(const v8h*)(brow + kk + 8 * g);
    v8h b1 = *(const v8h*)(brow + kk + 16 + 8 * g);
    v16h a = __builtin_shufflevector(a0, a1, 0, 1, 2, 3, 4, 5, 6, 7, 8, 9, 10, 11, 12, 13, 14, 15);
    v16h b = __builtin_shufflevector(b0, b1, 0, 1, 2, 3, 4, 5, 6, 7, 8, 9, 10, 11, 12, 13, 14, 15);
    acc = __builtin_amdgcn_wmma_f32_16x16x32_f16(false, a, false, b, (short)0, acc, false, false);
  }
  // C/D: VGPR r -> M = r + 8*g, N = lane&15
#pragma unroll
  for (int r = 0; r < 8; ++r)
    Cf[(size_t)(row0 + r + 8 * g) * FD + col0 + m] = acc[r];
}

// ------------- per-node attention scores s_src/s_dst -------------
template <int H, int C>
__global__ void k_scores(const float* __restrict__ h, const float* __restrict__ as,
                         const float* __restrict__ ad, float* __restrict__ ssrc,
                         float* __restrict__ sdst) {
  int idx = blockIdx.x * blockDim.x + threadIdx.x;
  if (idx >= NN * H) return;
  int n = idx / H, hh = idx % H;
  const float* hr = h + (size_t)n * FD + hh * C;
  const float* ar = as + hh * C;
  const float* br = ad + hh * C;
  float a = 0.f, b = 0.f;
#pragma unroll 8
  for (int c = 0; c < C; ++c) {
    float v = hr[c];
    a += v * ar[c];
    b += v * br[c];
  }
  ssrc[idx] = a;
  sdst[idx] = b;
}

__device__ __forceinline__ void edge_nodes(int t, const int* __restrict__ ei, int& s, int& d) {
  if (t < EE) { s = ei[t]; d = ei[EE + t]; } else { s = d = t - EE; }
}

// ------------- edge softmax: pass 1 segment max -------------
template <int H>
__global__ void k_edge_max(const int* __restrict__ ei, const float* __restrict__ ssrc,
                           const float* __restrict__ sdst, unsigned* __restrict__ mx) {
  int t = blockIdx.x * blockDim.x + threadIdx.x;
  if (t >= ET) return;
  int s, d;
  edge_nodes(t, ei, s, d);
#pragma unroll
  for (int hh = 0; hh < H; ++hh) {
    float al = ssrc[s * H + hh] + sdst[d * H + hh];
    al = al > 0.f ? al : 0.2f * al;  // leaky relu 0.2
    atomicMax(&mx[d * H + hh], fkey(al));
  }
}

// ------------- edge softmax: pass 2 exp + segment sum -------------
template <int H>
__global__ void k_edge_exp(const int* __restrict__ ei, const float* __restrict__ ssrc,
                           const float* __restrict__ sdst, const unsigned* __restrict__ mx,
                           float* __restrict__ dn, float* __restrict__ ev) {
  int t = blockIdx.x * blockDim.x + threadIdx.x;
  if (t >= ET) return;
  int s, d;
  edge_nodes(t, ei, s, d);
#pragma unroll
  for (int hh = 0; hh < H; ++hh) {
    float al = ssrc[s * H + hh] + sdst[d * H + hh];
    al = al > 0.f ? al : 0.2f * al;
    float m = funkey(mx[d * H + hh]);
    float e = expf(al - m);
    ev[(size_t)t * H + hh] = e;
    atomicAdd(&dn[d * H + hh], e);
  }
}

// ------------- aggregation: agg[dst] += (e/denom) * h[src]; one wave per edge -------------
template <int H>
__global__ void k_aggregate(const int* __restrict__ ei, const float* __restrict__ hfeat,
                            const float* __restrict__ ev, const float* __restrict__ dn,
                            float* __restrict__ agg) {
  int wave = threadIdx.x >> 5, lane = threadIdx.x & 31;
  int t = blockIdx.x * 8 + wave;
  if (t >= ET) return;
  int s, d;
  edge_nodes(t, ei, s, d);
  constexpr int C = FD / H;
#pragma unroll
  for (int q = 0; q < 4; ++q) {
    int j = lane + 32 * q;
    int hh = j / C;
    float coef = ev[(size_t)t * H + hh] / (dn[d * H + hh] + 1e-16f);
    atomicAdd(&agg[(size_t)d * FD + j], coef * hfeat[(size_t)s * FD + j]);
  }
}

// ------------- bias + LayerNorm + ReLU; one wave per node; also emits f16 copy -------------
__global__ void k_bias_ln_relu(const float* __restrict__ agg, const float* __restrict__ bias,
                               const float* __restrict__ gamma, const float* __restrict__ beta,
                               float* __restrict__ hout, _Float16* __restrict__ hh) {
  int wave = threadIdx.x >> 5, lane = threadIdx.x & 31;
  int n = blockIdx.x * 8 + wave;
  if (n >= NN) return;
  float v[4];
  float s = 0.f;
#pragma unroll
  for (int q = 0; q < 4; ++q) {
    int j = lane + 32 * q;
    v[q] = agg[(size_t)n * FD + j] + bias[j];
    s += v[q];
  }
#pragma unroll
  for (int o = 16; o > 0; o >>= 1) s += __shfl_xor(s, o, 32);
  float mu = s * (1.f / FD);
  float var = 0.f;
#pragma unroll
  for (int q = 0; q < 4; ++q) {
    float dd = v[q] - mu;
    var += dd * dd;
  }
#pragma unroll
  for (int o = 16; o > 0; o >>= 1) var += __shfl_xor(var, o, 32);
  float rs = rsqrtf(var * (1.f / FD) + 1e-5f);
#pragma unroll
  for (int q = 0; q < 4; ++q) {
    int j = lane + 32 * q;
    float y = (v[q] - mu) * rs * gamma[j] + beta[j];
    y = fmaxf(y, 0.f);
    hout[(size_t)n * FD + j] = y;
    hh[(size_t)n * FD + j] = (_Float16)y;
  }
}

// ------------- global mean pool -------------
__global__ void k_pool_acc(const float* __restrict__ h, const int* __restrict__ batch,
                           float* __restrict__ sums, float* __restrict__ cnt) {
  int wave = threadIdx.x >> 5, lane = threadIdx.x & 31;
  int n = blockIdx.x * 8 + wave;
  if (n >= NN) return;
  int g = batch[n];
#pragma unroll
  for (int q = 0; q < 4; ++q) {
    int j = lane + 32 * q;
    atomicAdd(&sums[(size_t)g * FD + j], h[(size_t)n * FD + j]);
  }
  if (lane == 0) atomicAdd(&cnt[g], 1.0f);
}

__global__ void k_pool_div(const float* __restrict__ sums, const float* __restrict__ cnt,
                           float* __restrict__ out) {
  int i = blockIdx.x * blockDim.x + threadIdx.x;
  if (i >= GG * FD) return;
  out[i] = sums[i] / fmaxf(cnt[i >> 7], 1.0f);
}

// ---------------- host-side edge phase driver (template dispatch) ----------------
static void run_edge_phase(int H, const int* ei, const float* ssrc, const float* sdst,
                           unsigned* mx, float* dn, float* ev, const float* hfeat, float* agg,
                           hipStream_t stream) {
  hipMemsetAsync(mx, 0, (size_t)NN * H * sizeof(unsigned), stream);  // 0 < fkey(-inf)
  hipMemsetAsync(dn, 0, (size_t)NN * H * sizeof(float), stream);
  hipMemsetAsync(agg, 0, (size_t)NN * FD * sizeof(float), stream);
  int eb = (ET + 255) / 256;
  int ab = (ET + 7) / 8;
  if (H == 4) {
    k_edge_max<4><<<eb, 256, 0, stream>>>(ei, ssrc, sdst, mx);
    k_edge_exp<4><<<eb, 256, 0, stream>>>(ei, ssrc, sdst, mx, dn, ev);
    k_aggregate<4><<<ab, 256, 0, stream>>>(ei, hfeat, ev, dn, agg);
  } else {
    k_edge_max<1><<<eb, 256, 0, stream>>>(ei, ssrc, sdst, mx);
    k_edge_exp<1><<<eb, 256, 0, stream>>>(ei, ssrc, sdst, mx, dn, ev);
    k_aggregate<1><<<ab, 256, 0, stream>>>(ei, hfeat, ev, dn, agg);
  }
}

extern "C" void kernel_launch(void* const* d_in, const int* in_sizes, int n_in,
                              void* d_out, int out_size, void* d_ws, size_t ws_size,
                              hipStream_t stream) {
  const float* x   = (const float*)d_in[0];
  const int* ei    = (const int*)d_in[1];
  const int* batch = (const int*)d_in[2];
  const float* W0  = (const float*)d_in[3];
  const float* as0 = (const float*)d_in[4];
  const float* ad0 = (const float*)d_in[5];
  const float* b0  = (const float*)d_in[6];
  const float* g0  = (const float*)d_in[7];
  const float* be0 = (const float*)d_in[8];
  const float* W1  = (const float*)d_in[9];
  const float* as1 = (const float*)d_in[10];
  const float* ad1 = (const float*)d_in[11];
  const float* b1  = (const float*)d_in[12];
  const float* g1  = (const float*)d_in[13];
  const float* be1 = (const float*)d_in[14];
  const float* W2  = (const float*)d_in[15];
  const float* as2 = (const float*)d_in[16];
  const float* ad2 = (const float*)d_in[17];
  const float* b2  = (const float*)d_in[18];
  const float* g2  = (const float*)d_in[19];
  const float* be2 = (const float*)d_in[20];

  // workspace carve-up (all offsets 256B aligned; total ~94 MB)
  char* ws      = (char*)d_ws;
  float* h      = (float*)(ws + 0);                 // [N,128] f32        25,600,000 B
  float* agg    = (float*)(ws + 25600000);          // [N,128] f32        25,600,000 B
  _Float16* hh  = (_Float16*)(ws + 51200000);       // [N,128] f16        12,800,000 B
  _Float16* Wt  = (_Float16*)(ws + 64000000);       // [128,128] f16          32,768 B
  float* ssrc   = (float*)(ws + 64032768);          // [N,4] f32             800,000 B
  float* sdst   = (float*)(ws + 64832768);          // [N,4] f32             800,000 B
  unsigned* mx  = (unsigned*)(ws + 65632768);       // [N,4] u32             800,000 B
  float* dn     = (float*)(ws + 66432768);          // [N,4] f32             800,000 B
  float* ev     = (float*)(ws + 67232768);          // [Et,4] f32         26,400,000 B
  float* psum   = (float*)(ws + 93632768);          // [G,128] f32           262,144 B
  float* pcnt   = (float*)(ws + 93894912);          // [G] f32                 2,048 B

  const int nb_node128 = (NN * FD + 255) / 256;
  const int nb_node    = (NN + 7) / 8;

  // ---- layer 0: GAT(4 -> 4x32 concat) ----
  k_lin0<<<nb_node128, 256, 0, stream>>>(x, W0, h);
  k_scores<4, 32><<<(NN * 4 + 255) / 256, 256, 0, stream>>>(h, as0, ad0, ssrc, sdst);
  run_edge_phase(4, ei, ssrc, sdst, mx, dn, ev, h, agg, stream);
  k_bias_ln_relu<<<nb_node, 256, 0, stream>>>(agg, b0, g0, be0, h, hh);

  // ---- layer 1: GAT(128 -> 4x32 concat), WMMA GEMM ----
  k_wT_half<<<(FD * FD + 255) / 256, 256, 0, stream>>>(W1, Wt);
  k_wmma_gemm<<<NN / 16, 256, 0, stream>>>(hh, Wt, h);
  k_scores<4, 32><<<(NN * 4 + 255) / 256, 256, 0, stream>>>(h, as1, ad1, ssrc, sdst);
  run_edge_phase(4, ei, ssrc, sdst, mx, dn, ev, h, agg, stream);
  k_bias_ln_relu<<<nb_node, 256, 0, stream>>>(agg, b1, g1, be1, h, hh);

  // ---- layer 2: GAT(128 -> 128, 1 head), WMMA GEMM ----
  k_wT_half<<<(FD * FD + 255) / 256, 256, 0, stream>>>(W2, Wt);
  k_wmma_gemm<<<NN / 16, 256, 0, stream>>>(hh, Wt, h);
  k_scores<1, 128><<<(NN + 255) / 256, 256, 0, stream>>>(h, as2, ad2, ssrc, sdst);
  run_edge_phase(1, ei, ssrc, sdst, mx, dn, ev, h, agg, stream);
  k_bias_ln_relu<<<nb_node, 256, 0, stream>>>(agg, b2, g2, be2, h, hh);

  // ---- global mean pool ----
  hipMemsetAsync(psum, 0, (size_t)GG * FD * sizeof(float), stream);
  hipMemsetAsync(pcnt, 0, (size_t)GG * sizeof(float), stream);
  k_pool_acc<<<nb_node, 256, 0, stream>>>(h, batch, psum, pcnt);
  k_pool_div<<<(GG * FD + 255) / 256, 256, 0, stream>>>(psum, pcnt, (float*)d_out);
}
